// SoftShapeNet_layer_6313601925435
// MI455X (gfx1250) — matrix operand order repaired and checked by reading, compiled, not run on hardware
//
#include <hip/hip_runtime.h>
#include <math.h>

// ---------------- problem constants ----------------
#define B_    8
#define S_    8192
#define D_    512
#define NF_   128
#define E_    4
#define HID_  512
#define L_    4096
#define T_    4097              // L+1 rows after token pruning
#define NTOK  32776             // B_*T_
#define NTOKP 32832             // padded to 64-token tiles (513*64)
#define TCH   513               // NTOKP/64
#define CTCH  65                // ceil(T_/64)
#define HALO  19
#define XBR   4224              // rows per batch in haloed xb buffer
#define SQRTD 22.62741699796952f
#define EPSIN 1e-5f
#define LDSROW 520              // padded LDS row (u16 units): 1040B = 260 dwords -> bank-conflict-free

typedef __bf16 bf16_t;
typedef bf16_t v16bf __attribute__((ext_vector_type(16)));
typedef float  v8f   __attribute__((ext_vector_type(8)));
typedef unsigned short u16;
typedef unsigned long long u64;

// ---------------- helpers ----------------
static __device__ __forceinline__ u16 f2bf(float f) {
    unsigned u = __float_as_uint(f);
    unsigned r = u + 0x7FFFu + ((u >> 16) & 1u);   // RNE
    return (u16)(r >> 16);
}
static __device__ __forceinline__ float bf2f(u16 h) {
    return __uint_as_float(((unsigned)h) << 16);
}
static __device__ __forceinline__ float gelu_exact(float x) {
    return 0.5f * x * (1.0f + erff(x * 0.7071067811865476f));
}

static __device__ __forceinline__ v8f wmma_bf16(v16bf a, v16bf b, v8f c) {
    // v_wmma_f32_16x16x32_bf16
    return __builtin_amdgcn_wmma_f32_16x16x32_bf16(false, a, false, b, (short)0, c, false, false);
}

// A fragment: 16x32 bf16 tile at (m0,k0) from row-major [M x ld] bf16-bit array.
// lane layout (ISA 7.12.2): lanes0-15 row=m0+lane, K = k0+[0..7], k0+16+[0..7]
//                           lanes16-31 row=m0+lane-16, K = k0+8+[0..7], k0+24+[0..7]
static __device__ __forceinline__ v16bf ldA(const u16* base, int ld, int m0, int k0, int lane) {
    int row  = m0 + (lane & 15);
    int half = lane >> 4;
    const u16* p = base + (size_t)row * ld + k0 + half * 8;
    v16bf a;
    uint4* ap = reinterpret_cast<uint4*>(&a);
    ap[0] = *reinterpret_cast<const uint4*>(p);        // K k0+half*8   .. +7
    ap[1] = *reinterpret_cast<const uint4*>(p + 16);   // K k0+16+half*8.. +7
    return a;
}
// B fragment: 32x16 tile (k0,n0); element(k,n) = rowmajor[n][k]  (token-major activations)
// lanes0-15: n=n0+lane, K=k0+[0..7],k0+16+[0..7] ; lanes16-31: n=n0+lane-16, K=k0+8+[0..7],k0+24+[0..7]
// loaded as two 16B chunks (works for global and LDS rows, only needs 16B alignment)
static __device__ __forceinline__ v16bf ldBrow(const u16* rowbase, int k0, int lane) {
    const u16* p = rowbase + k0 + (lane >> 4) * 8;
    v16bf b;
    uint4* bp = reinterpret_cast<uint4*>(&b);
    bp[0] = *reinterpret_cast<const uint4*>(p);
    bp[1] = *reinterpret_cast<const uint4*>(p + 16);
    return b;
}

// Cooperative async staging of a 64-token x 512 bf16 panel into LDS.
// Each thread issues 16x global_load_async_to_lds_b128 (ASYNCcnt-tracked DMA into LDS),
// then waits its own ASYNCcnt; caller must __syncthreads() before consuming.
static __device__ __forceinline__ void stage_panel_async(const u16* __restrict__ gsrc,
                                                         u16* __restrict__ smem,
                                                         int tid) {
    unsigned lbase = (unsigned)(size_t)(void*)smem;   // low 32 bits of LDS aperture = LDS byte offset
#pragma unroll
    for (int q = 0; q < 16; ++q) {
        int c   = tid + q * 256;       // 0..4095 : 16B chunk id
        int row = c >> 6;              // 0..63  token row
        int col = c & 63;              // 16B chunk within the 1024B row
        unsigned la = lbase + (unsigned)(row * (LDSROW * 2) + col * 16);
        unsigned long long ga = (unsigned long long)(size_t)(gsrc + (size_t)row * D_ + col * 8);
        asm volatile("global_load_async_to_lds_b128 %0, %1, off"
                     :: "v"(la), "v"(ga) : "memory");
    }
    asm volatile("s_wait_asynccnt 0x0" ::: "memory");
}

// ---------------- weight prep (f32 -> bf16, with transposes) ----------------
__global__ void k_f2bf(const float* __restrict__ in, u16* __restrict__ out, int n) {
    int i = blockIdx.x * 256 + threadIdx.x;
    if (i < n) out[i] = f2bf(in[i]);
}
__global__ void k_prep_w1(const float* __restrict__ W1, u16* __restrict__ W1t) {
    int i = blockIdx.x * 256 + threadIdx.x;          // over [e][h][d]
    if (i >= E_ * HID_ * D_) return;
    int d = i & 511, h = (i >> 9) & 511, e = i >> 18;
    W1t[i] = f2bf(W1[((size_t)e * D_ + d) * HID_ + h]);
}
__global__ void k_prep_w2(const float* __restrict__ W2, u16* __restrict__ W2t) {
    int i = blockIdx.x * 256 + threadIdx.x;          // over [e][d][h]
    if (i >= E_ * D_ * HID_) return;
    int h = i & 511, d = (i >> 9) & 511, e = i >> 18;
    W2t[i] = f2bf(W2[((size_t)e * HID_ + h) * D_ + d]);
}
__global__ void k_prep_conv(const float* __restrict__ Wc, u16* __restrict__ out, int taps) {
    int i = blockIdx.x * 256 + threadIdx.x;          // over [k][o][ic]
    if (i >= taps * NF_ * NF_) return;
    int ic = i & 127, o = (i >> 7) & 127, k = i >> 14;
    out[i] = f2bf(Wc[((size_t)o * NF_ + ic) * taps + k]);
}

// ---------------- norm1 + score + weighted tokens ----------------
__global__ __launch_bounds__(128) void k_norm1(const float* __restrict__ x,
                                               const float* __restrict__ gamma1,
                                               const float* __restrict__ Wa,
                                               const float* __restrict__ ba,
                                               float* __restrict__ xs,
                                               float* __restrict__ score) {
    int tok = blockIdx.x;
    int tid = threadIdx.x;
    size_t base = (size_t)tok * D_;
    float4 xv = *reinterpret_cast<const float4*>(x + base + tid * 4);
    float g0 = gamma1[tid*4], g1 = gamma1[tid*4+1], g2 = gamma1[tid*4+2], g3 = gamma1[tid*4+3];
    float a0 = Wa[tid*4], a1 = Wa[tid*4+1], a2 = Wa[tid*4+2], a3 = Wa[tid*4+3];
    float p2 = xv.x*xv.x + xv.y*xv.y + xv.z*xv.z + xv.w*xv.w;
    float pw = xv.x*g0*a0 + xv.y*g1*a1 + xv.z*g2*a2 + xv.w*g3*a3;
    __shared__ float s2[128], sw[128];
    s2[tid] = p2; sw[tid] = pw;
    __syncthreads();
    for (int s = 64; s > 0; s >>= 1) {
        if (tid < s) { s2[tid] += s2[tid + s]; sw[tid] += sw[tid + s]; }
        __syncthreads();
    }
    float n = fmaxf(sqrtf(s2[0]), 1e-12f);
    float scale = SQRTD / n;
    float z = sw[0] * scale + ba[0];
    float sc = 1.0f / (1.0f + expf(-z));
    if (tid == 0) score[tok] = sc;
    float m = scale * sc;
    float4 o; o.x = xv.x*g0*m; o.y = xv.y*g1*m; o.z = xv.z*g2*m; o.w = xv.w*g3*m;
    *reinterpret_cast<float4*>(xs + base + tid * 4) = o;
}

// ---------------- exact top-k (L = S/2) via bitonic sort in LDS ----------------
__global__ __launch_bounds__(1024) void k_topk(const float* __restrict__ score,
                                               int* __restrict__ sidx,
                                               char* __restrict__ sel,
                                               float* __restrict__ out_sidx_f) {
    __shared__ u64 keys[S_];                 // 64 KB of the WGP's 320 KB LDS
    int b = blockIdx.x, tid = threadIdx.x;
    for (int i = tid; i < S_; i += 1024) {
        float s = score[(size_t)b * S_ + i];
        unsigned fu = __float_as_uint(s);
        fu = (fu & 0x80000000u) ? ~fu : (fu | 0x80000000u);    // order-preserving
        keys[i] = ((u64)fu << 32) | (unsigned)(~(unsigned)i);  // score desc, idx asc
        sel[(size_t)b * S_ + i] = 0;
    }
    // descending bitonic sort of 8192 keys
    for (unsigned size = 2; size <= S_; size <<= 1)
        for (unsigned stride = size >> 1; stride >= 1; stride >>= 1) {
            __syncthreads();
            for (unsigned t = tid; t < S_ / 2; t += 1024) {
                unsigned i = 2 * t - (t & (stride - 1));
                unsigned j = i + stride;
                u64 a = keys[i], c = keys[j];
                bool desc = (i & size) == 0;
                if (desc ? (a < c) : (a > c)) { keys[i] = c; keys[j] = a; }
            }
        }
    __syncthreads();
    // extract top-L indices, mark selection mask
    unsigned myidx[4];
#pragma unroll
    for (int q = 0; q < 4; ++q) myidx[q] = ~(unsigned)(keys[tid + q * 1024] & 0xffffffffull);
    __syncthreads();
    unsigned* ib = reinterpret_cast<unsigned*>(keys);
#pragma unroll
    for (int q = 0; q < 4; ++q) {
        ib[tid + q * 1024] = myidx[q];
        sel[(size_t)b * S_ + myidx[q]] = 1;
    }
    // ascending bitonic sort of the L survivors -> sorted_left_idx
    for (unsigned size = 2; size <= L_; size <<= 1)
        for (unsigned stride = size >> 1; stride >= 1; stride >>= 1) {
            __syncthreads();
            for (unsigned t = tid; t < L_ / 2; t += 1024) {
                unsigned i = 2 * t - (t & (stride - 1));
                unsigned j = i + stride;
                unsigned a = ib[i], c = ib[j];
                bool asc = (i & size) == 0;
                if (asc ? (a > c) : (a < c)) { ib[i] = c; ib[j] = a; }
            }
        }
    __syncthreads();
#pragma unroll
    for (int q = 0; q < 4; ++q) {
        int i = tid + q * 1024;
        sidx[(size_t)b * L_ + i] = (int)ib[i];
        out_sidx_f[(size_t)b * L_ + i] = (float)ib[i];
    }
}

// ---------------- complement-sum ("extra" row) ----------------
__global__ __launch_bounds__(512) void k_extra(const float* __restrict__ xs,
                                               const char* __restrict__ sel,
                                               float* __restrict__ extra) {
    int b = blockIdx.x, chunk = blockIdx.y, d = threadIdx.x;
    float acc = 0.f;
    int s0 = chunk * 1024;
    for (int s = s0; s < s0 + 1024; ++s)
        if (!sel[(size_t)b * S_ + s]) acc += xs[((size_t)b * S_ + s) * D_ + d];
    atomicAdd(&extra[(size_t)b * D_ + d], acc);
}

// ---------------- gather pruned sequence xc = [left ; extra] ----------------
__global__ __launch_bounds__(128) void k_gather(const float* __restrict__ xs,
                                                const int* __restrict__ sidx,
                                                const float* __restrict__ extra,
                                                float* __restrict__ xc) {
    int blk = blockIdx.x;                 // 0..NTOK-1
    int b = blk / T_, i = blk - b * T_;
    const float* src = (i < L_) ? xs + ((size_t)b * S_ + sidx[(size_t)b * L_ + i]) * D_
                                : extra + (size_t)b * D_;
    float4 v = *reinterpret_cast<const float4*>(src + threadIdx.x * 4);
    *reinterpret_cast<float4*>(xc + (size_t)blk * D_ + threadIdx.x * 4) = v;
}

// ---------------- rmsnorm2 -> bf16 activations ----------------
__global__ __launch_bounds__(128) void k_norm2(const float* __restrict__ xc,
                                               const float* __restrict__ gamma2,
                                               u16* __restrict__ xn2) {
    int tok = blockIdx.x, tid = threadIdx.x;
    size_t base = (size_t)tok * D_;
    float4 xv = *reinterpret_cast<const float4*>(xc + base + tid * 4);
    __shared__ float s2[128];
    s2[tid] = xv.x*xv.x + xv.y*xv.y + xv.z*xv.z + xv.w*xv.w;
    __syncthreads();
    for (int s = 64; s > 0; s >>= 1) {
        if (tid < s) s2[tid] += s2[tid + s];
        __syncthreads();
    }
    float scale = SQRTD / fmaxf(sqrtf(s2[0]), 1e-12f);
    u16* o = xn2 + base + tid * 4;
    o[0] = f2bf(xv.x * gamma2[tid*4]   * scale);
    o[1] = f2bf(xv.y * gamma2[tid*4+1] * scale);
    o[2] = f2bf(xv.z * gamma2[tid*4+2] * scale);
    o[3] = f2bf(xv.w * gamma2[tid*4+3] * scale);
}

// ---------------- maxpool(k=3,s=1,p=1) along T, per batch ----------------
__global__ __launch_bounds__(128) void k_maxpool(const u16* __restrict__ xn2,
                                                 u16* __restrict__ mpT) {
    int tok = blockIdx.x;                 // 0..NTOK-1
    int b = tok / T_, pos = tok - b * T_;
    const u16* c  = xn2 + (size_t)tok * D_;
    const u16* pm = (pos > 0)      ? c - D_ : c;
    const u16* pp = (pos < T_ - 1) ? c + D_ : c;
    int d0 = threadIdx.x * 4;
#pragma unroll
    for (int q = 0; q < 4; ++q) {
        float v = fmaxf(bf2f(c[d0+q]), fmaxf(bf2f(pm[d0+q]), bf2f(pp[d0+q])));
        mpT[(size_t)tok * D_ + d0 + q] = f2bf(v);
    }
}

// ---------------- MoE gating + importance accumulation ----------------
__global__ __launch_bounds__(128) void k_gates(const u16* __restrict__ xn2,
                                               const float* __restrict__ Wg,
                                               float* __restrict__ gates,
                                               float* __restrict__ impsum) {
    int tok = blockIdx.x, tid = threadIdx.x;
    const u16* xr = xn2 + (size_t)tok * D_;
    float acc[E_] = {0.f, 0.f, 0.f, 0.f};
#pragma unroll
    for (int q = 0; q < 4; ++q) {
        int d = tid * 4 + q;
        float xv = bf2f(xr[d]);
#pragma unroll
        for (int e = 0; e < E_; ++e) acc[e] += xv * Wg[(size_t)d * E_ + e];
    }
    __shared__ float sg[128][E_];
#pragma unroll
    for (int e = 0; e < E_; ++e) sg[tid][e] = acc[e];
    __syncthreads();
    for (int s = 64; s > 0; s >>= 1) {
        if (tid < s)
#pragma unroll
            for (int e = 0; e < E_; ++e) sg[tid][e] += sg[tid + s][e];
        __syncthreads();
    }
    if (tid == 0) {
        float m = fmaxf(fmaxf(sg[0][0], sg[0][1]), fmaxf(sg[0][2], sg[0][3]));
        float ex[E_], sum = 0.f;
#pragma unroll
        for (int e = 0; e < E_; ++e) { ex[e] = expf(sg[0][e] - m); sum += ex[e]; }
        float inv = 1.0f / sum;
#pragma unroll
        for (int e = 0; e < E_; ++e) {
            float g = ex[e] * inv;
            gates[(size_t)tok * E_ + e] = g;
            atomicAdd(&impsum[e], g);
        }
    }
}

// ---------------- MoE GEMM1: async-stage B panel to LDS, H^T tile = W1t[e] x xn2^T ----------------
__global__ __launch_bounds__(256) void k_moe_gemm1(const u16* __restrict__ Wt,  // [HID][D]
                                                   const u16* __restrict__ X,   // [NTOKP][D]
                                                   u16* __restrict__ H) {       // [NTOKP][HID]
    __shared__ u16 sB[64][LDSROW];        // 65 KB staged activation panel
    int tid = threadIdx.x;
    int lane = tid & 31, w = tid >> 5;
    int m0 = blockIdx.y * 128 + w * 16;   // hidden dim
    int t0 = blockIdx.x * 64;             // tokens
    stage_panel_async(X + (size_t)t0 * D_, &sB[0][0], tid);
    __syncthreads();
    v8f z = {0.f,0.f,0.f,0.f,0.f,0.f,0.f,0.f};
    v8f acc[4] = {z, z, z, z};
    const u16* rp[4];
#pragma unroll
    for (int j = 0; j < 4; ++j) rp[j] = &sB[j * 16 + (lane & 15)][0];
#pragma unroll
    for (int k0 = 0; k0 < D_; k0 += 32) {
        v16bf a = ldA(Wt, D_, m0, k0, lane);
#pragma unroll
        for (int j = 0; j < 4; ++j) acc[j] = wmma_bf16(a, ldBrow(rp[j], k0, lane), acc[j]);
    }
    int n = lane & 15, half = lane >> 4;
#pragma unroll
    for (int j = 0; j < 4; ++j) {
        int t = t0 + j * 16 + n;
#pragma unroll
        for (int v = 0; v < 8; ++v)
            H[(size_t)t * HID_ + m0 + v + 8 * half] = f2bf(gelu_exact(acc[j][v]));
    }
}

// ---------------- MoE GEMM2: async-stage H panel, Y = W2t[e] x H^T; out += gate_e * Y ----------------
__global__ __launch_bounds__(256) void k_moe_gemm2(const u16* __restrict__ Wt,   // [D][HID]
                                                   const u16* __restrict__ H,    // [NTOKP][HID]
                                                   const float* __restrict__ gates,
                                                   float* __restrict__ moeacc,   // [NTOK][D]
                                                   int e) {
    __shared__ u16 sB[64][LDSROW];
    int tid = threadIdx.x;
    int lane = tid & 31, w = tid >> 5;
    int m0 = blockIdx.y * 128 + w * 16;   // output dim
    int t0 = blockIdx.x * 64;
    stage_panel_async(H + (size_t)t0 * HID_, &sB[0][0], tid);
    __syncthreads();
    v8f z = {0.f,0.f,0.f,0.f,0.f,0.f,0.f,0.f};
    v8f acc[4] = {z, z, z, z};
    const u16* rp[4];
#pragma unroll
    for (int j = 0; j < 4; ++j) rp[j] = &sB[j * 16 + (lane & 15)][0];
#pragma unroll
    for (int k0 = 0; k0 < HID_; k0 += 32) {
        v16bf a = ldA(Wt, HID_, m0, k0, lane);
#pragma unroll
        for (int j = 0; j < 4; ++j) acc[j] = wmma_bf16(a, ldBrow(rp[j], k0, lane), acc[j]);
    }
    int n = lane & 15, half = lane >> 4;
#pragma unroll
    for (int j = 0; j < 4; ++j) {
        int t = t0 + j * 16 + n;
        if (t < NTOK) {
            float g = gates[(size_t)t * E_ + e];
#pragma unroll
            for (int v = 0; v < 8; ++v) {
                size_t o = (size_t)t * D_ + m0 + v + 8 * half;
                moeacc[o] += g * acc[j][v];
            }
        }
    }
}

// ---------------- 1x1 bottleneck: xb = Wb x xn2^T -> haloed token-major xbT ----------------
__global__ __launch_bounds__(256) void k_xb_gemm(const u16* __restrict__ Wbb,   // [NF][D]
                                                 const u16* __restrict__ X,     // xn2 [NTOKP][D]
                                                 const u16* __restrict__ zrow,
                                                 u16* __restrict__ xbT) {       // [B][XBR][NF]
    int lane = threadIdx.x & 31, w = threadIdx.x >> 5;
    int b = blockIdx.z;
    int m0 = w * 16;                      // out channel
    int t0 = blockIdx.x * 64;
    v8f z = {0.f,0.f,0.f,0.f,0.f,0.f,0.f,0.f};
    v8f acc[4] = {z, z, z, z};
    const u16* rp[4];
#pragma unroll
    for (int j = 0; j < 4; ++j) {
        int t = t0 + j * 16 + (lane & 15);
        rp[j] = (t < T_) ? X + (size_t)(b * T_ + t) * D_ : zrow;
    }
#pragma unroll
    for (int k0 = 0; k0 < D_; k0 += 32) {
        v16bf a = ldA(Wbb, D_, m0, k0, lane);
#pragma unroll
        for (int j = 0; j < 4; ++j) acc[j] = wmma_bf16(a, ldBrow(rp[j], k0, lane), acc[j]);
    }
    int n = lane & 15, half = lane >> 4;
#pragma unroll
    for (int j = 0; j < 4; ++j) {
        int t = t0 + j * 16 + n;
        if (t < T_) {
#pragma unroll
            for (int v = 0; v < 8; ++v)
                xbT[((size_t)b * XBR + HALO + t) * NF_ + m0 + v + 8 * half] = f2bf(acc[j][v]);
        }
    }
}

// ---------------- maxpool branch 1x1: y[384..511] = Wm x mp^T ----------------
__global__ __launch_bounds__(256) void k_mp_gemm(const u16* __restrict__ Wmb,   // [NF][D]
                                                 const u16* __restrict__ MP,    // [NTOKP][D]
                                                 const u16* __restrict__ zrow,
                                                 float* __restrict__ y) {       // [B][512][T]
    int lane = threadIdx.x & 31, w = threadIdx.x >> 5;
    int b = blockIdx.z;
    int m0 = w * 16;
    int t0 = blockIdx.x * 64;
    v8f z = {0.f,0.f,0.f,0.f,0.f,0.f,0.f,0.f};
    v8f acc[4] = {z, z, z, z};
    const u16* rp[4];
#pragma unroll
    for (int j = 0; j < 4; ++j) {
        int t = t0 + j * 16 + (lane & 15);
        rp[j] = (t < T_) ? MP + (size_t)(b * T_ + t) * D_ : zrow;
    }
#pragma unroll
    for (int k0 = 0; k0 < D_; k0 += 32) {
        v16bf a = ldA(Wmb, D_, m0, k0, lane);
#pragma unroll
        for (int j = 0; j < 4; ++j) acc[j] = wmma_bf16(a, ldBrow(rp[j], k0, lane), acc[j]);
    }
    int n = lane & 15, half = lane >> 4;
#pragma unroll
    for (int j = 0; j < 4; ++j) {
        int t = t0 + j * 16 + n;
        if (t < T_) {
#pragma unroll
            for (int v = 0; v < 8; ++v)
                y[((size_t)b * 512 + 384 + m0 + v + 8 * half) * T_ + t] = acc[j][v];
        }
    }
}

// ---------------- tap-decomposed conv1d as WMMA GEMM ----------------
__global__ __launch_bounds__(256) void k_conv(const u16* __restrict__ Wck,  // [taps][NF][NF]
                                              const u16* __restrict__ xbT,  // [B][XBR][NF]
                                              float* __restrict__ y,        // [B][512][T]
                                              int taps, int P, int cOff) {
    int lane = threadIdx.x & 31, w = threadIdx.x >> 5;
    int b = blockIdx.z;
    int m0 = w * 16;
    int t0 = blockIdx.x * 64;
    const u16* xb = xbT + (size_t)b * XBR * NF_;
    int n = lane & 15, half = lane >> 4;
    int r0[4];
#pragma unroll
    for (int j = 0; j < 4; ++j) r0[j] = HALO + t0 + j * 16 + n - P;
    v8f z = {0.f,0.f,0.f,0.f,0.f,0.f,0.f,0.f};
    v8f acc[4] = {z, z, z, z};
    for (int k = 0; k < taps; ++k) {
        const u16* wk = Wck + (size_t)k * NF_ * NF_;
#pragma unroll
        for (int c = 0; c < 4; ++c) {
            int k0 = c * 32;
            v16bf a = ldA(wk, NF_, m0, k0, lane);
#pragma unroll
            for (int j = 0; j < 4; ++j) {
                const u16* rowp = xb + (size_t)(r0[j] + k) * NF_;
                acc[j] = wmma_bf16(a, ldBrow(rowp, k0, lane), acc[j]);
            }
        }
    }
#pragma unroll
    for (int j = 0; j < 4; ++j) {
        int t = t0 + j * 16 + n;
        if (t < T_) {
#pragma unroll
            for (int v = 0; v < 8; ++v)
                y[((size_t)b * 512 + cOff + m0 + v + 8 * half) * T_ + t] = acc[j][v];
        }
    }
}

// ---------------- InstanceNorm stats over T per (b, channel) ----------------
__global__ __launch_bounds__(256) void k_instnorm(const float* __restrict__ y,
                                                  float* __restrict__ mu,
                                                  float* __restrict__ rs) {
    int cl = blockIdx.x;                   // b*512 + c
    int tid = threadIdx.x;
    const float* row = y + (size_t)cl * T_;
    float s = 0.f, q = 0.f;
    for (int t = tid; t < T_; t += 256) { float v = row[t]; s += v; q += v * v; }
    __shared__ float ss[256], sq[256];
    ss[tid] = s; sq[tid] = q;
    __syncthreads();
    for (int st = 128; st > 0; st >>= 1) {
        if (tid < st) { ss[tid] += ss[tid + st]; sq[tid] += sq[tid + st]; }
        __syncthreads();
    }
    if (tid == 0) {
        float m = ss[0] / (float)T_;
        float var = sq[0] / (float)T_ - m * m;
        mu[cl] = m;
        rs[cl] = rsqrtf(var + EPSIN);
    }
}

// ---------------- final fusion: gelu(xc + moe + gelu(instnorm(conv))) ----------------
__global__ __launch_bounds__(128) void k_final(const float* __restrict__ xc,
                                               const float* __restrict__ moeacc,
                                               const float* __restrict__ y,
                                               const float* __restrict__ mu,
                                               const float* __restrict__ rs,
                                               float* __restrict__ out) {
    int blk = blockIdx.x;                  // token row 0..NTOK-1
    int b = blk / T_, i = blk - b * T_;
    int c0 = threadIdx.x * 4;
#pragma unroll
    for (int q = 0; q < 4; ++q) {
        int c = c0 + q;
        int cl = b * 512 + c;
        float yv  = y[(size_t)cl * T_ + i];
        float inc = gelu_exact((yv - mu[cl]) * rs[cl]);
        float v = xc[(size_t)blk * D_ + c] + moeacc[(size_t)blk * D_ + c] + inc;
        out[(size_t)blk * D_ + c] = gelu_exact(v);
    }
}

// ---------------- moe load-balance loss ----------------
__global__ void k_loss(const float* __restrict__ impsum, float* __restrict__ out_loss) {
    float loss = 0.f;
#pragma unroll
    for (int e = 0; e < E_; ++e) {
        float imp = impsum[e] / (float)NTOK;
        loss += imp * imp;
    }
    out_loss[0] = (float)E_ * loss;
}

// ================= host side =================
extern "C" void kernel_launch(void* const* d_in, const int* in_sizes, int n_in,
                              void* d_out, int out_size, void* d_ws, size_t ws_size,
                              hipStream_t stream) {
    (void)in_sizes; (void)n_in; (void)out_size; (void)ws_size;
    const float* x      = (const float*)d_in[0];
    const float* gamma1 = (const float*)d_in[1];
    const float* gamma2 = (const float*)d_in[2];
    const float* Wa     = (const float*)d_in[3];
    const float* ba     = (const float*)d_in[4];
    const float* Wb     = (const float*)d_in[5];
    const float* Wc39   = (const float*)d_in[6];
    const float* Wc19   = (const float*)d_in[7];
    const float* Wc9    = (const float*)d_in[8];
    const float* Wm     = (const float*)d_in[9];
    const float* Wg     = (const float*)d_in[10];
    const float* W1     = (const float*)d_in[11];
    const float* W2     = (const float*)d_in[12];
    float* out = (float*)d_out;
    const size_t OUT_MAIN = (size_t)B_ * T_ * D_;     // 16781312
    float* out_loss = out + OUT_MAIN;
    float* out_sidx = out + OUT_MAIN + 1;

    char* ws = (char*)d_ws;
    size_t off = 0;
    auto take = [&](size_t bytes) -> char* {
        char* p = ws + off;
        off += (bytes + 255) & ~(size_t)255;
        return p;
    };
    float* xs     = (float*)take((size_t)B_ * S_ * D_ * 4);
    float* score  = (float*)take((size_t)B_ * S_ * 4);
    int*   sidx   = (int*)  take((size_t)B_ * L_ * 4);
    char*  sel    = (char*) take((size_t)B_ * S_);
    float* extra  = (float*)take((size_t)B_ * D_ * 4);
    float* xc     = (float*)take((size_t)NTOK * D_ * 4);
    u16*   xn2    = (u16*)  take((size_t)NTOKP * D_ * 2);
    u16*   mpT    = (u16*)  take((size_t)NTOKP * D_ * 2);
    u16*   Hbuf   = (u16*)  take((size_t)NTOKP * HID_ * 2);
    float* moeacc = (float*)take((size_t)NTOK * D_ * 4);
    float* gates  = (float*)take((size_t)NTOK * E_ * 4);
    float* impsum = (float*)take(256);
    float* ybuf   = (float*)take((size_t)B_ * 512 * T_ * 4);
    float* mu     = (float*)take(4096 * 4);
    float* rs     = (float*)take(4096 * 4);
    u16*   xbT    = (u16*)  take((size_t)B_ * XBR * NF_ * 2);
    u16*   zrow   = (u16*)  take((size_t)D_ * 2);
    u16*   w1t    = (u16*)  take((size_t)E_ * HID_ * D_ * 2);
    u16*   w2t    = (u16*)  take((size_t)E_ * D_ * HID_ * 2);
    u16*   wbb    = (u16*)  take((size_t)NF_ * D_ * 2);
    u16*   wmb    = (u16*)  take((size_t)NF_ * D_ * 2);
    u16*   wc39k  = (u16*)  take((size_t)39 * NF_ * NF_ * 2);
    u16*   wc19k  = (u16*)  take((size_t)19 * NF_ * NF_ * 2);
    u16*   wc9k   = (u16*)  take((size_t)9 * NF_ * NF_ * 2);

    // per-call zero init (deterministic; capture-safe)
    hipMemsetAsync(extra, 0, (size_t)B_ * D_ * 4, stream);
    hipMemsetAsync(impsum, 0, 256, stream);
    hipMemsetAsync(moeacc, 0, (size_t)NTOK * D_ * 4, stream);
    hipMemsetAsync(xbT, 0, (size_t)B_ * XBR * NF_ * 2, stream);
    hipMemsetAsync(zrow, 0, (size_t)D_ * 2, stream);
    hipMemsetAsync(xn2 + (size_t)NTOK * D_, 0, (size_t)(NTOKP - NTOK) * D_ * 2, stream);
    hipMemsetAsync(mpT + (size_t)NTOK * D_, 0, (size_t)(NTOKP - NTOK) * D_ * 2, stream);

    // weight prep (fp32 -> bf16, transposed for contiguous WMMA A-fragments)
    k_f2bf<<<(NF_ * D_ + 255) / 256, 256, 0, stream>>>(Wb, wbb, NF_ * D_);
    k_f2bf<<<(NF_ * D_ + 255) / 256, 256, 0, stream>>>(Wm, wmb, NF_ * D_);
    k_prep_w1<<<(E_ * HID_ * D_ + 255) / 256, 256, 0, stream>>>(W1, w1t);
    k_prep_w2<<<(E_ * D_ * HID_ + 255) / 256, 256, 0, stream>>>(W2, w2t);
    k_prep_conv<<<(39 * NF_ * NF_ + 255) / 256, 256, 0, stream>>>(Wc39, wc39k, 39);
    k_prep_conv<<<(19 * NF_ * NF_ + 255) / 256, 256, 0, stream>>>(Wc19, wc19k, 19);
    k_prep_conv<<<(9 * NF_ * NF_ + 255) / 256, 256, 0, stream>>>(Wc9, wc9k, 9);

    // pipeline
    k_norm1<<<B_ * S_, 128, 0, stream>>>(x, gamma1, Wa, ba, xs, score);
    k_topk<<<B_, 1024, 0, stream>>>(score, sidx, sel, out_sidx);
    k_extra<<<dim3(B_, S_ / 1024), 512, 0, stream>>>(xs, sel, extra);
    k_gather<<<NTOK, 128, 0, stream>>>(xs, sidx, extra, xc);
    k_norm2<<<NTOK, 128, 0, stream>>>(xc, gamma2, xn2);
    k_maxpool<<<NTOK, 128, 0, stream>>>(xn2, mpT);
    k_gates<<<NTOK, 128, 0, stream>>>(xn2, Wg, gates, impsum);

    // inception branches (all WMMA)
    k_xb_gemm<<<dim3(CTCH, 1, B_), 256, 0, stream>>>(wbb, xn2, zrow, xbT);
    k_mp_gemm<<<dim3(CTCH, 1, B_), 256, 0, stream>>>(wmb, mpT, zrow, ybuf);
    k_conv<<<dim3(CTCH, 1, B_), 256, 0, stream>>>(wc39k, xbT, ybuf, 39, 19, 0);
    k_conv<<<dim3(CTCH, 1, B_), 256, 0, stream>>>(wc19k, xbT, ybuf, 19, 9, 128);
    k_conv<<<dim3(CTCH, 1, B_), 256, 0, stream>>>(wc9k, xbT, ybuf, 9, 4, 256);

    // MoE experts, sequential accumulate (WMMA + async-LDS staging)
    for (int e = 0; e < E_; ++e) {
        k_moe_gemm1<<<dim3(TCH, 4), 256, 0, stream>>>(w1t + (size_t)e * HID_ * D_, xn2, Hbuf);
        k_moe_gemm2<<<dim3(TCH, 4), 256, 0, stream>>>(w2t + (size_t)e * D_ * HID_, Hbuf, gates, moeacc, e);
    }

    k_instnorm<<<B_ * 512, 256, 0, stream>>>(ybuf, mu, rs);
    k_final<<<NTOK, 128, 0, stream>>>(xc, moeacc, ybuf, mu, rs, out);
    k_loss<<<1, 1, 0, stream>>>(impsum, out_loss);
}
